// Spatial_Attention_conv_48627619725504
// MI455X (gfx1250) — matrix operand order repaired and checked by reading, not compiled, not run on hardware
//
#include <hip/hip_runtime.h>
#include <hip/hip_bf16.h>

#define C_DIM 256
#define L_DIM 8192
#define D_DIM 128
#define OUT_DIM 256
#define EPSF 1e-5f

typedef __attribute__((ext_vector_type(16))) __bf16 v16bf;
typedef __attribute__((ext_vector_type(8)))  float  v8f;

// ---------------------------------------------------------------------------
// Kernel 1: per-channel GroupNorm statistics (mean, rsqrt(var+eps)) over L.
// ---------------------------------------------------------------------------
__global__ __launch_bounds__(256) void gn_stats_kernel(const float* __restrict__ x,
                                                       float* __restrict__ mean,
                                                       float* __restrict__ rstd) {
    __shared__ float s1[256], s2[256];
    const int c = blockIdx.x, t = threadIdx.x;
    const float* xc = x + (size_t)c * L_DIM;
    float a = 0.f, b = 0.f;
    for (int k = 0; k < L_DIM; k += 256) {
        float v = xc[k + t];
        a += v;
        b += v * v;
    }
    s1[t] = a; s2[t] = b;
    __syncthreads();
    for (int off = 128; off > 0; off >>= 1) {
        if (t < off) { s1[t] += s1[t + off]; s2[t] += s2[t + off]; }
        __syncthreads();
    }
    if (t == 0) {
        float m   = s1[0] * (1.0f / L_DIM);
        float var = s2[0] * (1.0f / L_DIM) - m * m;
        mean[c] = m;
        rstd[c] = rsqrtf(var + EPSF);
    }
}

// ---------------------------------------------------------------------------
// Kernel 2: fused GroupNorm + 1x1 conv + LeakyReLU for q/k/v, writing bf16
// operands pre-swizzled into WMMA A / B VGPR layouts.
//   rows 0..127  -> q  (B operand of GEMM1: K=d, N=j)
//   rows 128..255-> k  (A operand of GEMM1: M=i, K=d)
//   rows 256..511-> v  (B operand of GEMM2: K=j, N=o)
// Grid: 512 blocks = 16 row-tiles(32 rows) x 32 col-tiles(256 cols).
// ---------------------------------------------------------------------------
__global__ __launch_bounds__(256) void proj_kernel(
    const float* __restrict__ x, const float* __restrict__ mean, const float* __restrict__ rstd,
    const float* __restrict__ gqw, const float* __restrict__ gqb,
    const float* __restrict__ wq,  const float* __restrict__ bq,
    const float* __restrict__ gkw, const float* __restrict__ gkb,
    const float* __restrict__ wk,  const float* __restrict__ bk,
    const float* __restrict__ gvw, const float* __restrict__ gvb,
    const float* __restrict__ wv,  const float* __restrict__ bv,
    __bf16* __restrict__ qB, __bf16* __restrict__ kA, __bf16* __restrict__ vB) {
    __shared__ float a_sh[32][256];
    __shared__ float off_sh[32];

    const int rt = blockIdx.x >> 5;        // row tile: 0..15
    const int lt = blockIdx.x & 31;        // col tile: 0..31
    const int t  = threadIdx.x;
    const int l  = lt * 256 + t;

    const float* W; const float* gw; const float* gb; const float* bias; int row0, kind;
    if (rt < 4)      { kind = 0; row0 = rt * 32;       W = wq; gw = gqw; gb = gqb; bias = bq; }
    else if (rt < 8) { kind = 1; row0 = rt * 32 - 128; W = wk; gw = gkw; gb = gkb; bias = bk; }
    else             { kind = 2; row0 = rt * 32 - 256; W = wv; gw = gvw; gb = gvb; bias = bv; }

    // Fold GN scale into per-channel coefficients a[i][c] = W[r][c]*rstd[c]*gw[c].
    {
        const float rs = rstd[t] * gw[t];
        #pragma unroll
        for (int i = 0; i < 32; ++i)
            a_sh[i][t] = W[(size_t)(row0 + i) * C_DIM + t] * rs;
    }
    // Per-row constant offset: bias + sum_c W*(gb - mean*rstd*gw).
    if (t < 32) {
        float o = bias[row0 + t];
        for (int c = 0; c < C_DIM; ++c) {
            float rs = rstd[c] * gw[c];
            o += W[(size_t)(row0 + t) * C_DIM + c] * (gb[c] - mean[c] * rs);
        }
        off_sh[t] = o;
    }
    __syncthreads();

    float acc[32];
    #pragma unroll
    for (int i = 0; i < 32; ++i) acc[i] = off_sh[i];

    for (int c = 0; c < C_DIM; ++c) {
        const float xv = x[(size_t)c * L_DIM + l];
        #pragma unroll
        for (int i = 0; i < 32; ++i)
            acc[i] = fmaf(a_sh[i][c], xv, acc[i]);
    }

    #pragma unroll
    for (int i = 0; i < 32; ++i) {
        float h = acc[i];
        h = (h >= 0.f) ? h : 0.1f * h;             // LeakyReLU(0.1)
        __bf16 hb = (__bf16)h;
        const int r = row0 + i;
        if (kind == 0) {            // q: B operand, K=d=r, N=j=l
            int jt = l >> 4, n = l & 15, kb = r >> 5, kp = r & 31;
            int lane = n + ((kp >> 4) << 4);
            int e = kp & 15;
            qB[(((size_t)(jt * 4 + kb)) * 32 + lane) * 16 + e] = hb;
        } else if (kind == 1) {     // k: A operand, M=i=l, K=d=r
            int it = l >> 4, M = l & 15, kb = r >> 5, kp = r & 31;
            int hi = kp >> 4;
            int e  = kp & 15;
            int lane = M + (hi << 4);
            kA[(((size_t)(it * 4 + kb)) * 32 + lane) * 16 + e] = hb;
        } else {                    // v: B operand, K=j=l, N=o=r
            int ot = r >> 4, n = r & 15, kb = l >> 5, kp = l & 31;
            int lane = n + ((kp >> 4) << 4);
            int e = kp & 15;
            vB[(((size_t)(ot * (L_DIM / 32) + kb)) * 32 + lane) * 16 + e] = hb;
        }
    }
}

// ---------------------------------------------------------------------------
// Kernel 3: GEMM1 raw scores S = key_v @ query_v (8192x8192, K=128), bf16 WMMA.
// Wave computes a 16x64 tile (A reused across 4 B tiles). 8 waves/block.
// ---------------------------------------------------------------------------
__global__ __launch_bounds__(256) void gemm1_kernel(const __bf16* __restrict__ kA,
                                                    const __bf16* __restrict__ qB,
                                                    float* __restrict__ atten) {
    const int wid  = threadIdx.x >> 5;
    const int lane = threadIdx.x & 31;
    const int task = blockIdx.x * 8 + wid;   // 65536 tasks
    const int it = task >> 7;                // i tile (512)
    const int jg = task & 127;               // j group of 64 (128)

    const v16bf* A = (const v16bf*)kA;
    const v16bf* B = (const v16bf*)qB;

    v8f acc[4] = {};
    #pragma unroll
    for (int kb = 0; kb < 4; ++kb) {
        v16bf a = A[((size_t)(it * 4 + kb)) * 32 + lane];
        #pragma unroll
        for (int tt = 0; tt < 4; ++tt) {
            v16bf b = B[((size_t)((jg * 4 + tt) * 4 + kb)) * 32 + lane];
            acc[tt] = __builtin_amdgcn_wmma_f32_16x16x32_bf16(
                false, a, false, b, (short)0, acc[tt], false, false);
        }
    }
    const int rbase = it * 16 + ((lane >> 4) << 3);
    const int cbase = jg * 64 + (lane & 15);
    #pragma unroll
    for (int tt = 0; tt < 4; ++tt)
        #pragma unroll
        for (int r = 0; r < 8; ++r)
            atten[(size_t)(rbase + r) * L_DIM + cbase + tt * 16] = acc[tt][r];
}

// ---------------------------------------------------------------------------
// Kernel 4: row softmax over 8192 columns, in place. One block per row,
// full row in registers (32 f32 per thread).
// ---------------------------------------------------------------------------
__global__ __launch_bounds__(256) void softmax_kernel(float* __restrict__ atten) {
    __shared__ float red[256];
    const int row = blockIdx.x, t = threadIdx.x;
    float4* pv = (float4*)(atten + (size_t)row * L_DIM + t * 32);

    float4 v[8];
    float mx = -3.0e38f;
    #pragma unroll
    for (int i = 0; i < 8; ++i) {
        v[i] = pv[i];
        mx = fmaxf(mx, fmaxf(fmaxf(v[i].x, v[i].y), fmaxf(v[i].z, v[i].w)));
    }
    red[t] = mx; __syncthreads();
    for (int off = 128; off > 0; off >>= 1) {
        if (t < off) red[t] = fmaxf(red[t], red[t + off]);
        __syncthreads();
    }
    mx = red[0]; __syncthreads();

    float s = 0.f;
    #pragma unroll
    for (int i = 0; i < 8; ++i) {
        v[i].x = __expf(v[i].x - mx); s += v[i].x;
        v[i].y = __expf(v[i].y - mx); s += v[i].y;
        v[i].z = __expf(v[i].z - mx); s += v[i].z;
        v[i].w = __expf(v[i].w - mx); s += v[i].w;
    }
    red[t] = s; __syncthreads();
    for (int off = 128; off > 0; off >>= 1) {
        if (t < off) red[t] += red[t + off];
        __syncthreads();
    }
    const float inv = 1.0f / red[0];
    #pragma unroll
    for (int i = 0; i < 8; ++i) {
        v[i].x *= inv; v[i].y *= inv; v[i].z *= inv; v[i].w *= inv;
        pv[i] = v[i];
    }
}

// ---------------------------------------------------------------------------
// Kernel 5: GEMM2 reweight = atten @ val_v (8192x256, K=8192), bf16 WMMA.
// A (f32 atten) converted to bf16 on the fly in A-lane layout; B pre-swizzled.
// Wave computes a 16x64 tile. 8 waves/block, 256 blocks.
// ---------------------------------------------------------------------------
__global__ __launch_bounds__(256) void gemm2_kernel(const float* __restrict__ atten,
                                                    const __bf16* __restrict__ vB,
                                                    float* __restrict__ outw) {
    const int wid  = threadIdx.x >> 5;
    const int lane = threadIdx.x & 31;
    const int task = blockIdx.x * 8 + wid;   // 2048 tasks
    const int it = task >> 2;                // i tile (512)
    const int og = task & 3;                 // o group of 64 (4)

    const int hi   = lane >> 4;
    const int mrow = it * 16 + (lane & 15);
    const float* arow = atten + (size_t)mrow * L_DIM + hi * 16;
    const v16bf* B = (const v16bf*)vB;

    v8f acc[4] = {};
    for (int kb = 0; kb < L_DIM / 32; ++kb) {
        const float4* ap = (const float4*)(arow + kb * 32);
        float4 q0 = ap[0], q1 = ap[1], q2 = ap[2], q3 = ap[3];
        v16bf a;
        a[0]  = (__bf16)q0.x; a[1]  = (__bf16)q0.y; a[2]  = (__bf16)q0.z; a[3]  = (__bf16)q0.w;
        a[4]  = (__bf16)q1.x; a[5]  = (__bf16)q1.y; a[6]  = (__bf16)q1.z; a[7]  = (__bf16)q1.w;
        a[8]  = (__bf16)q2.x; a[9]  = (__bf16)q2.y; a[10] = (__bf16)q2.z; a[11] = (__bf16)q2.w;
        a[12] = (__bf16)q3.x; a[13] = (__bf16)q3.y; a[14] = (__bf16)q3.z; a[15] = (__bf16)q3.w;
        #pragma unroll
        for (int tt = 0; tt < 4; ++tt) {
            v16bf b = B[((size_t)((og * 4 + tt) * (L_DIM / 32) + kb)) * 32 + lane];
            acc[tt] = __builtin_amdgcn_wmma_f32_16x16x32_bf16(
                false, a, false, b, (short)0, acc[tt], false, false);
        }
    }
    const int rbase = it * 16 + (hi << 3);
    const int cbase = og * 64 + (lane & 15);
    #pragma unroll
    for (int tt = 0; tt < 4; ++tt)
        #pragma unroll
        for (int r = 0; r < 8; ++r)
            outw[(size_t)(rbase + r) * OUT_DIM + cbase + tt * 16] = acc[tt][r];
}

// ---------------------------------------------------------------------------
extern "C" void kernel_launch(void* const* d_in, const int* in_sizes, int n_in,
                              void* d_out, int out_size, void* d_ws, size_t ws_size,
                              hipStream_t stream) {
    const float* x   = (const float*)d_in[0];
    const float* gqw = (const float*)d_in[1];
    const float* gqb = (const float*)d_in[2];
    const float* wq  = (const float*)d_in[3];
    const float* bq  = (const float*)d_in[4];
    const float* gkw = (const float*)d_in[5];
    const float* gkb = (const float*)d_in[6];
    const float* wk  = (const float*)d_in[7];
    const float* bk  = (const float*)d_in[8];
    const float* gvw = (const float*)d_in[9];
    const float* gvb = (const float*)d_in[10];
    const float* wv  = (const float*)d_in[11];
    const float* bv  = (const float*)d_in[12];

    float* outw  = (float*)d_out;                                // (L, OUT)
    float* atten = outw + (size_t)L_DIM * OUT_DIM;               // (L, L)

    char* ws = (char*)d_ws;
    float*  mean = (float*)ws;
    float*  rstd = (float*)(ws + 1024);
    __bf16* qB   = (__bf16*)(ws + 4096);
    __bf16* kA   = qB + (size_t)D_DIM * L_DIM;
    __bf16* vB   = kA + (size_t)D_DIM * L_DIM;

    gn_stats_kernel<<<C_DIM, 256, 0, stream>>>(x, mean, rstd);
    proj_kernel<<<512, 256, 0, stream>>>(x, mean, rstd,
                                         gqw, gqb, wq, bq,
                                         gkw, gkb, wk, bk,
                                         gvw, gvb, wv, bv,
                                         qB, kA, vB);
    gemm1_kernel<<<8192, 256, 0, stream>>>(kA, qB, atten);
    softmax_kernel<<<L_DIM, 256, 0, stream>>>(atten);
    gemm2_kernel<<<256, 256, 0, stream>>>(atten, vB, outw);
}
